// StructuredStateSpace_38130719654300
// MI455X (gfx1250) — matrix operand order repaired and checked by reading, not compiled, not run on hardware
//
#include <hip/hip_runtime.h>
#include <hip/hip_bf16.h>
#include <stdint.h>

// Problem constants (match reference)
#define S_LEN   8192
#define NBATCH  16
#define DST     128      // state dim
#define DIN     64       // in/out feature dim
#define CHUNK   128      // blocked-scan chunk length
#define NCHUNK  (S_LEN / CHUNK)   // 64
#define HSTR    136      // padded LDS row stride (bf16 elems): 272B -> +4 bank shift/row

typedef __bf16 bf16t;
typedef __attribute__((ext_vector_type(16))) __bf16 v16bf;
typedef __attribute__((ext_vector_type(8)))  __bf16 v8bf;
typedef __attribute__((ext_vector_type(8)))  float  v8f;

union V16U { v16bf v; v8bf h[2]; __bf16 e[16]; };

static __device__ __forceinline__ v8f wmma_bf16(v16bf a, v16bf b, v8f c) {
  // D = A(16x32 bf16) x B(32x16 bf16) + C(16x16 f32)
  return __builtin_amdgcn_wmma_f32_16x16x32_bf16(false, a, false, b, (short)0, c,
                                                 false, false);
}

// Split workgroup barrier: producer side waits for its own LDS traffic to
// retire, then signals; latency-tolerant work goes between signal and wait.
static __device__ __forceinline__ void wg_barrier_signal() {
  asm volatile("s_wait_dscnt 0\n"
               "s_barrier_signal -1" ::: "memory");
}
static __device__ __forceinline__ void wg_barrier_wait() {
  asm volatile("s_barrier_wait -1" ::: "memory");
}

// A-operand (16x32 bf16) chunk c from LDS tile, row-major [16][HSTR] bf16.
// Lanes 0-15 hold row M=lane, K in {0..15}+32c;
// lanes 16-31 hold row M=lane-16, K in {16..31}+32c.
static __device__ __forceinline__ v16bf load_aop(const bf16t* h, int lane, int c) {
  const int m  = lane & 15;
  const int k0 = c * 32 + ((lane & 16) ? 16 : 0);
  V16U u;
  u.h[0] = *(const v8bf*)(h + m * HSTR + k0);
  u.h[1] = *(const v8bf*)(h + m * HSTR + k0 + 8);
  return u.v;
}

// B-operand (32x16 bf16) chunk c, N-tile nt, from global row-major [N][ldk] bf16.
// Lanes 0-15: N=lane, K=32c..32c+15 ; lanes 16-31: N=lane-16, K=32c+16..32c+31.
static __device__ __forceinline__ v16bf load_bop(const bf16t* B, int lane, int nt,
                                                 int c, int ldk) {
  const int n  = nt * 16 + (lane & 15);
  const int k0 = c * 32 + ((lane & 16) ? 16 : 0);
  return *(const v16bf*)(B + (size_t)n * ldk + k0);
}

// ---------------------------------------------------------------- prep kernels
__global__ void k_cvt_bf16(const float* __restrict__ s, bf16t* __restrict__ d, int n) {
  int i = blockIdx.x * blockDim.x + threadIdx.x;
  int st = gridDim.x * blockDim.x;
  for (; i < n; i += st) d[i] = (bf16t)s[i];
}

// out = in @ in (128x128 f32). Used 7x (repeated squaring) to form A^128 in f32.
__global__ void k_matsq(const float* __restrict__ in, float* __restrict__ out) {
  __shared__ float row[DST];
  const int m = blockIdx.x, n = threadIdx.x;   // blockDim.x == 128
  row[n] = in[m * DST + n];
  __syncthreads();
  float acc = 0.f;
#pragma unroll 8
  for (int k = 0; k < DST; ++k) acc = fmaf(row[k], in[k * DST + n], acc);
  out[m * DST + n] = acc;
}

// --------------------------------------------------- phase 1: bx = x@Bw^T + Bb
// grid (S/16, BATCH), block 256. Wave w handles N-tile w (8 tiles of 16).
__global__ void __launch_bounds__(256)
k_inproj(const float* __restrict__ x, const bf16t* __restrict__ Bwbf,
         const float* __restrict__ Bb, float* __restrict__ bx) {
  const int t0   = blockIdx.x * 16;
  const int b    = blockIdx.y;
  const int w    = threadIdx.x >> 5;
  const int lane = threadIdx.x & 31;
  const int n    = w * 16 + (lane & 15);
  const int hi   = (lane >> 4) & 1;

  const v16bf bw0 = load_bop(Bwbf, lane, w, 0, DIN);
  const v16bf bw1 = load_bop(Bwbf, lane, w, 1, DIN);

  // A-operand rows are time steps t0..t0+15; K = DIN = 64 (2 chunks)
  const int m  = lane & 15;
  const int kb = (lane & 16) ? 16 : 0;
  const float* xp = x + ((size_t)b * S_LEN + (t0 + m)) * DIN;
  V16U a0, a1;
#pragma unroll
  for (int e = 0; e < 8; ++e) {
    a0.e[e]     = (bf16t)xp[kb + e];
    a0.e[8 + e] = (bf16t)xp[kb + 8 + e];
    a1.e[e]     = (bf16t)xp[32 + kb + e];
    a1.e[8 + e] = (bf16t)xp[32 + kb + 8 + e];
  }

  const float bias = Bb[n];
  v8f c;
#pragma unroll
  for (int r = 0; r < 8; ++r) c[r] = bias;

  v8f d = wmma_bf16(a0.v, bw0, c);
  d = wmma_bf16(a1.v, bw1, d);

#pragma unroll
  for (int r = 0; r < 8; ++r) {
    const int tm = t0 + r + 8 * hi;
    bx[((size_t)tm * NBATCH + b) * DST + n] = d[r];
  }
}

// ------------------------------------- phase 2a: chunk-local scan (zero init)
// grid NCHUNK, block 256 (8 waves). Keeps only chunk-final state L_c (f32).
// Double-buffered LDS state + split barrier: one barrier per step, next step's
// bx loads issued in the barrier shadow.
__global__ void __launch_bounds__(256)
k_scan_local(const float* __restrict__ bx, const bf16t* __restrict__ Abf,
             float* __restrict__ wsL) {
  __shared__ alignas(32) bf16t hsm[2][NBATCH * HSTR];
  const int c    = blockIdx.x;
  const int w    = threadIdx.x >> 5;
  const int lane = threadIdx.x & 31;
  const int n    = w * 16 + (lane & 15);
  const int hi   = (lane >> 4) & 1;
  const int t0   = c * CHUNK;

  v16bf At[4];
#pragma unroll
  for (int k = 0; k < 4; ++k) At[k] = load_bop(Abf, lane, w, k, DST);

  for (int i = threadIdx.x; i < NBATCH * DST; i += 256)
    hsm[0][(i >> 7) * HSTR + (i & 127)] = (bf16t)0.f;
  __syncthreads();

  // preload C tile (bx) for first step
  v8f cc;
  {
    const float* bp = bx + (size_t)t0 * (NBATCH * DST) + hi * 8 * DST + n;
#pragma unroll
    for (int r = 0; r < 8; ++r) cc[r] = bp[r * DST];
  }

  int p = 0;
  v8f d = {0.f, 0.f, 0.f, 0.f, 0.f, 0.f, 0.f, 0.f};
  for (int it = 0; it < CHUNK; ++it) {
    const v16bf a0 = load_aop(hsm[p], lane, 0);
    const v16bf a1 = load_aop(hsm[p], lane, 1);
    const v16bf a2 = load_aop(hsm[p], lane, 2);
    const v16bf a3 = load_aop(hsm[p], lane, 3);
    d = wmma_bf16(a0, At[0], cc);
    d = wmma_bf16(a1, At[1], d);
    d = wmma_bf16(a2, At[2], d);
    d = wmma_bf16(a3, At[3], d);
#pragma unroll
    for (int r = 0; r < 8; ++r) hsm[p ^ 1][(r + 8 * hi) * HSTR + n] = (bf16t)d[r];
    wg_barrier_signal();
    // ---- barrier shadow: fetch next step's C tile ----
    v8f ccn = cc;
    if (it + 1 < CHUNK) {
      const float* bp = bx + (size_t)(t0 + it + 1) * (NBATCH * DST) + hi * 8 * DST + n;
#pragma unroll
      for (int r = 0; r < 8; ++r) ccn[r] = bp[r * DST];
      if (it + 3 < CHUNK)
        __builtin_prefetch(bx + (size_t)(t0 + it + 3) * (NBATCH * DST) + hi * 8 * DST + n, 0, 1);
    }
    wg_barrier_wait();
    cc = ccn;
    p ^= 1;
  }
  float* lp = wsL + (size_t)c * (NBATCH * DST);
#pragma unroll
  for (int r = 0; r < 8; ++r) lp[(r + 8 * hi) * DST + n] = d[r];
}

// --------------------- phase 2b: sequential combine over chunks (1 workgroup)
// carry_{c+1} = carry_c @ (A^128)^T + L_c ; stores carry_c (bf16) for phase 2c.
__global__ void __launch_bounds__(256)
k_scan_combine(const float* __restrict__ wsL, const bf16t* __restrict__ ALbf,
               bf16t* __restrict__ wsCy) {
  __shared__ alignas(32) bf16t hsm[2][NBATCH * HSTR];
  const int w    = threadIdx.x >> 5;
  const int lane = threadIdx.x & 31;
  const int n    = w * 16 + (lane & 15);
  const int hi   = (lane >> 4) & 1;

  v16bf At[4];
#pragma unroll
  for (int k = 0; k < 4; ++k) At[k] = load_bop(ALbf, lane, w, k, DST);

  for (int i = threadIdx.x; i < NBATCH * DST; i += 256)
    hsm[0][(i >> 7) * HSTR + (i & 127)] = (bf16t)0.f;
  __syncthreads();

  v8f cc;
  {
    const float* lp = wsL + hi * 8 * DST + n;
#pragma unroll
    for (int r = 0; r < 8; ++r) cc[r] = lp[r * DST];
  }

  int p = 0;
  for (int c = 0; c < NCHUNK; ++c) {
    // snapshot carry_c for the emit pass
    for (int i = threadIdx.x; i < NBATCH * DST; i += 256)
      wsCy[(size_t)c * (NBATCH * DST) + i] = hsm[p][(i >> 7) * HSTR + (i & 127)];
    const v16bf a0 = load_aop(hsm[p], lane, 0);
    const v16bf a1 = load_aop(hsm[p], lane, 1);
    const v16bf a2 = load_aop(hsm[p], lane, 2);
    const v16bf a3 = load_aop(hsm[p], lane, 3);
    v8f d = wmma_bf16(a0, At[0], cc);
    d = wmma_bf16(a1, At[1], d);
    d = wmma_bf16(a2, At[2], d);
    d = wmma_bf16(a3, At[3], d);
#pragma unroll
    for (int r = 0; r < 8; ++r) hsm[p ^ 1][(r + 8 * hi) * HSTR + n] = (bf16t)d[r];
    wg_barrier_signal();
    v8f ccn = cc;
    if (c + 1 < NCHUNK) {
      const float* lp = wsL + (size_t)(c + 1) * (NBATCH * DST) + hi * 8 * DST + n;
#pragma unroll
      for (int r = 0; r < 8; ++r) ccn[r] = lp[r * DST];
    }
    wg_barrier_wait();
    cc = ccn;
    p ^= 1;
  }
}

// ------------- phase 2c: re-scan with correct carry; fused y_t = h_t @ Cw^T
// y projection WMMAs + stores run in the barrier shadow (off critical path).
__global__ void __launch_bounds__(256)
k_scan_emit(const float* __restrict__ bx, const bf16t* __restrict__ Abf,
            const bf16t* __restrict__ Cwbf, const bf16t* __restrict__ wsCy,
            float* __restrict__ y) {
  __shared__ alignas(32) bf16t hsm[2][NBATCH * HSTR];
  const int c    = blockIdx.x;
  const int w    = threadIdx.x >> 5;
  const int lane = threadIdx.x & 31;
  const int n    = w * 16 + (lane & 15);
  const int hi   = (lane >> 4) & 1;
  const int t0   = c * CHUNK;

  v16bf At[4];
#pragma unroll
  for (int k = 0; k < 4; ++k) At[k] = load_bop(Abf, lane, w, k, DST);
  v16bf Ct[4];
  if (w < 4) {
#pragma unroll
    for (int k = 0; k < 4; ++k) Ct[k] = load_bop(Cwbf, lane, w, k, DST);
  }

  for (int i = threadIdx.x; i < NBATCH * DST; i += 256)
    hsm[0][(i >> 7) * HSTR + (i & 127)] = wsCy[(size_t)c * (NBATCH * DST) + i];
  __syncthreads();

  v8f cc;
  {
    const float* bp = bx + (size_t)t0 * (NBATCH * DST) + hi * 8 * DST + n;
#pragma unroll
    for (int r = 0; r < 8; ++r) cc[r] = bp[r * DST];
  }

  int p = 0;
  for (int it = 0; it < CHUNK; ++it) {
    const int t = t0 + it;
    const v16bf a0 = load_aop(hsm[p], lane, 0);
    const v16bf a1 = load_aop(hsm[p], lane, 1);
    const v16bf a2 = load_aop(hsm[p], lane, 2);
    const v16bf a3 = load_aop(hsm[p], lane, 3);
    v8f d = wmma_bf16(a0, At[0], cc);
    d = wmma_bf16(a1, At[1], d);
    d = wmma_bf16(a2, At[2], d);
    d = wmma_bf16(a3, At[3], d);
#pragma unroll
    for (int r = 0; r < 8; ++r) hsm[p ^ 1][(r + 8 * hi) * HSTR + n] = (bf16t)d[r];
    wg_barrier_signal();
    // ---- barrier shadow: y projection for h_{t-1} + next C tile ----
    if (w < 4 && it > 0) {
      v8f z = {0.f, 0.f, 0.f, 0.f, 0.f, 0.f, 0.f, 0.f};
      v8f yd = wmma_bf16(a0, Ct[0], z);
      yd = wmma_bf16(a1, Ct[1], yd);
      yd = wmma_bf16(a2, Ct[2], yd);
      yd = wmma_bf16(a3, Ct[3], yd);
      const int tm = t - 1;
#pragma unroll
      for (int r = 0; r < 8; ++r)
        y[((size_t)(r + 8 * hi) * S_LEN + tm) * DIN + n] = yd[r];
    }
    v8f ccn = cc;
    if (it + 1 < CHUNK) {
      const float* bp = bx + (size_t)(t + 1) * (NBATCH * DST) + hi * 8 * DST + n;
#pragma unroll
      for (int r = 0; r < 8; ++r) ccn[r] = bp[r * DST];
      if (it + 3 < CHUNK)
        __builtin_prefetch(bx + (size_t)(t + 3) * (NBATCH * DST) + hi * 8 * DST + n, 0, 1);
    }
    wg_barrier_wait();
    cc = ccn;
    p ^= 1;
  }
  // tail: y for the chunk's last step (state is in hsm[p] after the loop)
  if (w < 4) {
    const v16bf a0 = load_aop(hsm[p], lane, 0);
    const v16bf a1 = load_aop(hsm[p], lane, 1);
    const v16bf a2 = load_aop(hsm[p], lane, 2);
    const v16bf a3 = load_aop(hsm[p], lane, 3);
    v8f z = {0.f, 0.f, 0.f, 0.f, 0.f, 0.f, 0.f, 0.f};
    v8f yd = wmma_bf16(a0, Ct[0], z);
    yd = wmma_bf16(a1, Ct[1], yd);
    yd = wmma_bf16(a2, Ct[2], yd);
    yd = wmma_bf16(a3, Ct[3], yd);
    const int tl = t0 + CHUNK - 1;
#pragma unroll
    for (int r = 0; r < 8; ++r)
      y[((size_t)(r + 8 * hi) * S_LEN + tl) * DIN + n] = yd[r];
  }
}

// ------------------------------------------------------------------- launcher
extern "C" void kernel_launch(void* const* d_in, const int* in_sizes, int n_in,
                              void* d_out, int out_size, void* d_ws, size_t ws_size,
                              hipStream_t stream) {
  (void)in_sizes; (void)n_in; (void)out_size; (void)ws_size;
  const float* x  = (const float*)d_in[0];
  const float* A  = (const float*)d_in[1];
  const float* Bw = (const float*)d_in[2];
  const float* Bb = (const float*)d_in[3];
  const float* Cw = (const float*)d_in[4];
  float* y = (float*)d_out;

  char* ws = (char*)d_ws;
  size_t off = 0;
  auto alloc = [&](size_t bytes) {
    off = (off + 255) & ~(size_t)255;
    void* p = ws + off;
    off += bytes;
    return p;
  };
  float* bx   = (float*)alloc((size_t)S_LEN * NBATCH * DST * sizeof(float)); // 64 MiB
  bf16t* Abf  = (bf16t*)alloc((size_t)DST * DST * sizeof(bf16t));
  bf16t* Bwbf = (bf16t*)alloc((size_t)DST * DIN * sizeof(bf16t));
  bf16t* Cwbf = (bf16t*)alloc((size_t)DIN * DST * sizeof(bf16t));
  bf16t* ALbf = (bf16t*)alloc((size_t)DST * DST * sizeof(bf16t));
  float* P0   = (float*)alloc((size_t)DST * DST * sizeof(float));
  float* P1   = (float*)alloc((size_t)DST * DST * sizeof(float));
  float* wsL  = (float*)alloc((size_t)NCHUNK * NBATCH * DST * sizeof(float));
  bf16t* wsCy = (bf16t*)alloc((size_t)NCHUNK * NBATCH * DST * sizeof(bf16t));

  // bf16 copies of weights (row-major [N][K] == B-operand friendly)
  k_cvt_bf16<<<32, 256, 0, stream>>>(A,  Abf,  DST * DST);
  k_cvt_bf16<<<16, 256, 0, stream>>>(Bw, Bwbf, DST * DIN);
  k_cvt_bf16<<<16, 256, 0, stream>>>(Cw, Cwbf, DIN * DST);

  // A^128 via 7 f32 squarings (keeps the carry chain accurate)
  k_matsq<<<DST, DST, 0, stream>>>(A,  P1);   // A^2
  k_matsq<<<DST, DST, 0, stream>>>(P1, P0);   // A^4
  k_matsq<<<DST, DST, 0, stream>>>(P0, P1);   // A^8
  k_matsq<<<DST, DST, 0, stream>>>(P1, P0);   // A^16
  k_matsq<<<DST, DST, 0, stream>>>(P0, P1);   // A^32
  k_matsq<<<DST, DST, 0, stream>>>(P1, P0);   // A^64
  k_matsq<<<DST, DST, 0, stream>>>(P0, P1);   // A^128
  k_cvt_bf16<<<32, 256, 0, stream>>>(P1, ALbf, DST * DST);

  dim3 g1(S_LEN / 16, NBATCH);
  k_inproj<<<g1, 256, 0, stream>>>(x, Bwbf, Bb, bx);

  k_scan_local<<<NCHUNK, 256, 0, stream>>>(bx, Abf, wsL);
  k_scan_combine<<<1, 256, 0, stream>>>(wsL, ALbf, wsCy);
  k_scan_emit<<<NCHUNK, 256, 0, stream>>>(bx, Abf, Cwbf, wsCy, y);
}
